// graph_directed_sep_init_18184891531338
// MI455X (gfx1250) — matrix operand order repaired and checked by reading, not compiled, not run on hardware
//
#include <hip/hip_runtime.h>

#define N_MOD 2
#define N_SUB 4096
#define DIM   256
#define TOPK  32
#define NNODES (N_MOD * N_SUB)   // 8192

typedef __attribute__((ext_vector_type(16))) __bf16 v16bf;
typedef __attribute__((ext_vector_type(8)))  __bf16 v8bf;
typedef __attribute__((ext_vector_type(8)))  float  v8f;

union bfvec { v16bf v; unsigned short s[16]; };

// concat two 8-lane bf16 vectors -> 16-lane (two b128 loads -> one WMMA operand)
__device__ __forceinline__ v16bf cat8(v8bf lo, v8bf hi) {
    return __builtin_shufflevector(lo, hi, 0, 1, 2, 3, 4, 5, 6, 7,
                                           8, 9, 10, 11, 12, 13, 14, 15);
}

// round-to-nearest-even f32 -> bf16 (bit pattern)
__device__ __forceinline__ unsigned short f2bf(float f) {
    unsigned u = __float_as_uint(f);
    unsigned r = u + 0x7FFFu + ((u >> 16) & 1u);
    return (unsigned short)(r >> 16);
}

// monotone key: larger float -> larger unsigned
__device__ __forceinline__ unsigned mkey(float f) {
    unsigned u = __float_as_uint(f);
    return (u & 0x80000000u) ? ~u : (u | 0x80000000u);
}

// ---------------------------------------------------------------------------
// Kernel 1: v = emb @ W^T + b   (per matrix g). One wave -> 16(M) x 64(N).
// A = emb[g] (4096x256 f32, row-major), B(k,n) = W[g][n][k].
// Output: bf16, row-major [g][4096][256].
// ---------------------------------------------------------------------------
__global__ __launch_bounds__(32)
void feat_gemm(const float* __restrict__ emb, const float* __restrict__ W,
               const float* __restrict__ bias, unsigned short* __restrict__ dst) {
    const int g    = blockIdx.z;
    const int m0   = blockIdx.y * 16;
    const int n0   = blockIdx.x * 64;
    const int lane = threadIdx.x;
    const int r    = lane & 15;
    const int h    = lane >> 4;           // lane half selects K sub-range

    const float* A  = emb + (size_t)g * N_SUB * DIM;
    const float* Bm = W   + (size_t)g * DIM * DIM;

    v8f acc[4] = {};
    for (int k0 = 0; k0 < DIM; k0 += 32) {
        // A frag 16x32: halves 0..7 = K[k0+8h ..], halves 8..15 = +16
        bfvec a;
        const float* ap = A + (size_t)(m0 + r) * DIM + k0 + h * 8;
#pragma unroll
        for (int t = 0; t < 8; ++t) { a.s[t] = f2bf(ap[t]); a.s[8 + t] = f2bf(ap[16 + t]); }
#pragma unroll
        for (int j = 0; j < 4; ++j) {
            // B frag 32x16: halves 0..7 = K[k0+8h ..], halves 8..15 = +16, N = r
            bfvec b;
            const float* bp = Bm + (size_t)(n0 + 16 * j + r) * DIM + k0 + h * 8;
#pragma unroll
            for (int t = 0; t < 16; ++t) b.s[t] = f2bf(bp[t < 8 ? t : t + 8]);
            acc[j] = __builtin_amdgcn_wmma_f32_16x16x32_bf16(
                false, a.v, false, b.v, (short)0, acc[j], false, false);
        }
    }
#pragma unroll
    for (int j = 0; j < 4; ++j) {
        const int   col = n0 + 16 * j + r;
        const float bv  = bias[g * DIM + col];
#pragma unroll
        for (int vr = 0; vr < 8; ++vr) {
            const int m = m0 + vr + 8 * h;     // C layout: VGPR vr -> row vr + 8*half
            dst[((size_t)g * N_SUB + m) * DIM + col] = f2bf(acc[j][vr] + bv);
        }
    }
}

// ---------------------------------------------------------------------------
// Kernel 2: blocks[g] = v1[g] @ v2[g]^T, written into assembled 8192x8192 adj.
// One wave -> 64x64 tile: 4 A-frags x 4 B-frags = 16 WMMAs per K step,
// all operands loaded as b128 vectors (no repack VALU in the hot loop).
// ---------------------------------------------------------------------------
__global__ __launch_bounds__(32)
void block_gemm(const __bf16* __restrict__ v1,
                const __bf16* __restrict__ v2,
                float* __restrict__ out) {
    const int g    = blockIdx.z;
    const int mb   = blockIdx.y;          // 64-row block (0..63)
    const int nb   = blockIdx.x;          // 64-col block (0..63)
    const int lane = threadIdx.x;
    const int r    = lane & 15;
    const int h    = lane >> 4;

    const __bf16* Vr = v1 + (size_t)g * N_SUB * DIM;
    const __bf16* Vc = v2 + (size_t)g * N_SUB * DIM;

    v8f acc[4][4] = {};
    for (int k0 = 0; k0 < DIM; k0 += 32) {
        v16bf a[4], b[4];
#pragma unroll
        for (int i = 0; i < 4; ++i) {
            const __bf16* ap = Vr + (size_t)(mb * 64 + 16 * i + r) * DIM + k0 + h * 8;
            __builtin_prefetch(ap + 32, 0, 1);      // next K chunk -> global_prefetch_b8
            a[i] = cat8(*(const v8bf*)ap, *(const v8bf*)(ap + 16));
        }
#pragma unroll
        for (int j = 0; j < 4; ++j) {
            const __bf16* bp = Vc + (size_t)(nb * 64 + 16 * j + r) * DIM + k0 + h * 8;
            b[j] = cat8(*(const v8bf*)bp, *(const v8bf*)(bp + 16));
        }
#pragma unroll
        for (int i = 0; i < 4; ++i)
#pragma unroll
            for (int j = 0; j < 4; ++j)
                acc[i][j] = __builtin_amdgcn_wmma_f32_16x16x32_bf16(
                    false, a[i], false, b[j], (short)0, acc[i][j], false, false);
    }

    // adj block (imod, jmod): g = imod * N_MOD + jmod
    const int imod = g >> 1, jmod = g & 1;
#pragma unroll
    for (int i = 0; i < 4; ++i)
#pragma unroll
        for (int j = 0; j < 4; ++j)
#pragma unroll
            for (int vr = 0; vr < 8; ++vr) {
                const int arow = imod * N_SUB + mb * 64 + 16 * i + vr + 8 * h;
                const int acol = jmod * N_SUB + nb * 64 + 16 * j + r;
                out[(size_t)arow * NNODES + acol] = acc[i][j][vr];
            }
}

// ---------------------------------------------------------------------------
// Kernel 3: in-place row-wise top-32 mask over adj[8192][8192].
// 256 threads/row; each thread keeps its 32 stride-256 elements in registers.
// Exact Kth-largest key via 3-level radix select (11+11+10 bits), then mask.
// ---------------------------------------------------------------------------
__global__ __launch_bounds__(256)
void topk_mask(float* __restrict__ adj) {
    __shared__ unsigned hist[2048];
    __shared__ unsigned s_bin, s_quota, s_tie;

    const int row = blockIdx.x;
    const int tid = threadIdx.x;
    float* rp = adj + (size_t)row * NNODES;

    float v[32];
#pragma unroll
    for (int t = 0; t < 32; ++t) v[t] = rp[tid + t * 256];

    // ---- level 1: bits [31:21] ----
    for (int i = tid; i < 2048; i += 256) hist[i] = 0;
    __syncthreads();
#pragma unroll
    for (int t = 0; t < 32; ++t) atomicAdd(&hist[mkey(v[t]) >> 21], 1u);
    __syncthreads();
    if (tid == 0) {
        unsigned cum = 0; int b = 2047;
        for (; b > 0; --b) { unsigned c = hist[b]; if (cum + c >= (unsigned)TOPK) break; cum += c; }
        s_bin = (unsigned)b; s_quota = (unsigned)TOPK - cum;
    }
    __syncthreads();
    const unsigned b1 = s_bin;
    __syncthreads();

    // ---- level 2: bits [20:10] ----
    for (int i = tid; i < 2048; i += 256) hist[i] = 0;
    __syncthreads();
#pragma unroll
    for (int t = 0; t < 32; ++t) {
        unsigned k = mkey(v[t]);
        if ((k >> 21) == b1) atomicAdd(&hist[(k >> 10) & 2047u], 1u);
    }
    __syncthreads();
    if (tid == 0) {
        unsigned cum = 0; int b = 2047;
        for (; b > 0; --b) { unsigned c = hist[b]; if (cum + c >= s_quota) break; cum += c; }
        s_bin = (b1 << 11) | (unsigned)b;      // 22-bit prefix
        s_quota = s_quota - cum;
    }
    __syncthreads();
    const unsigned p2 = s_bin;
    __syncthreads();

    // ---- level 3: bits [9:0] ----
    for (int i = tid; i < 1024; i += 256) hist[i] = 0;
    __syncthreads();
#pragma unroll
    for (int t = 0; t < 32; ++t) {
        unsigned k = mkey(v[t]);
        if ((k >> 10) == p2) atomicAdd(&hist[k & 1023u], 1u);
    }
    __syncthreads();
    if (tid == 0) {
        unsigned cum = 0; int b = 1023;
        for (; b > 0; --b) { unsigned c = hist[b]; if (cum + c >= s_quota) break; cum += c; }
        s_bin = (p2 << 10) | (unsigned)b;      // exact Kth-largest key T
        s_quota = s_quota - cum;               // #ties (== T) to keep
        s_tie = 0;
    }
    __syncthreads();
    const unsigned T = s_bin;
    const unsigned E = s_quota;

#pragma unroll
    for (int t = 0; t < 32; ++t) {
        const unsigned k = mkey(v[t]);
        float o = 0.0f;
        if (k > T) o = v[t];
        else if (k == T) { unsigned p = atomicAdd(&s_tie, 1u); if (p < E) o = v[t]; }
        rp[tid + t * 256] = o;
    }
}

// ---------------------------------------------------------------------------
extern "C" void kernel_launch(void* const* d_in, const int* in_sizes, int n_in,
                              void* d_out, int out_size, void* d_ws, size_t ws_size,
                              hipStream_t stream) {
    // inputs: idx, emb1, emb2, W1, b1, W2, b2
    const float* emb1 = (const float*)d_in[1];
    const float* emb2 = (const float*)d_in[2];
    const float* W1   = (const float*)d_in[3];
    const float* b1   = (const float*)d_in[4];
    const float* W2   = (const float*)d_in[5];
    const float* b2   = (const float*)d_in[6];
    float* out = (float*)d_out;

    unsigned short* v1s = (unsigned short*)d_ws;                // 4*4096*256 bf16 = 8 MB
    unsigned short* v2s = v1s + (size_t)4 * N_SUB * DIM;        // +8 MB

    dim3 g1(DIM / 64, N_SUB / 16, 4);      // (4, 256, 4), 32 thr
    feat_gemm<<<g1, 32, 0, stream>>>(emb1, W1, b1, v1s);
    feat_gemm<<<g1, 32, 0, stream>>>(emb2, W2, b2, v2s);

    dim3 g2(N_SUB / 64, N_SUB / 64, 4);    // (64, 64, 4), 32 thr
    block_gemm<<<g2, 32, 0, stream>>>((const __bf16*)v1s, (const __bf16*)v2s, out);

    topk_mask<<<NNODES, 256, 0, stream>>>(out);
}
